// stMMD_loss_65541200937317
// MI455X (gfx1250) — compile-verified, matrix operand order checked by reading, not run on hardware
//
#include <hip/hip_runtime.h>
#include <hip/hip_bf16.h>

typedef __attribute__((ext_vector_type(16))) __bf16 bf16x16;
typedef __attribute__((ext_vector_type(8)))  __bf16 bf16x8;
typedef __attribute__((ext_vector_type(8)))  float  f32x8;

#define NROW 4096
#define NDIM 512
#define TM   128
#define TN   64
#define KSTEP 32

// Load a 16-element bf16 A/B fragment slice for this lane.
// Per ISA layout (16-bit A 16x32): lanes 0-15 hold K {0..7,16..23},
// lanes 16-31 hold K {8..15,24..31}; caller passes pointer already at
// row*D + kb + half*8; the second 8 elements sit +16 along K.
__device__ __forceinline__ bf16x16 load_frag(const __bf16* p) {
    bf16x8 a = *(const bf16x8*)p;
    bf16x8 b = *(const bf16x8*)(p + 16);
    return __builtin_shufflevector(a, b, 0,1,2,3,4,5,6,7,8,9,10,11,12,13,14,15);
}

// ---------------- K1: squared norms + bf16 hi/lo split ----------------
__global__ void k_prep(const float* __restrict__ X, __bf16* __restrict__ Xhi,
                       __bf16* __restrict__ Xlo, float* __restrict__ sq) {
    const int row = blockIdx.x;
    const int tid = threadIdx.x;
    __shared__ float red[256];
    float acc = 0.f;
    for (int e = tid; e < NDIM; e += 256) {
        float x = X[(size_t)row * NDIM + e];
        __bf16 h = (__bf16)x;
        __bf16 l = (__bf16)(x - (float)h);
        Xhi[(size_t)row * NDIM + e] = h;
        Xlo[(size_t)row * NDIM + e] = l;
        acc += x * x;
    }
    red[tid] = acc;
    __syncthreads();
    for (int s = 128; s > 0; s >>= 1) {
        if (tid < s) red[tid] += red[tid + s];
        __syncthreads();
    }
    if (tid == 0) sq[row] = red[0];
}

// ---------------- K2: WMMA GEMM tile -> L2 + per-tile sum ----------------
// grid = (NROW/TM, NROW/TN), block = 256 (8 wave32). Wave grid 4x2:
// each wave computes a 32x32 patch = 2x2 fragments of 16x16.
__global__ void k_gemm(const __bf16* __restrict__ Xhi, const __bf16* __restrict__ Xlo,
                       const float* __restrict__ sq, float* __restrict__ L2,
                       float* __restrict__ partial) {
    const int wave  = threadIdx.x >> 5;
    const int lane  = threadIdx.x & 31;
    const int half  = lane >> 4;
    const int ln    = lane & 15;
    const int waveM = wave & 3;
    const int waveN = wave >> 2;
    const int rBase = blockIdx.x * TM + waveM * 32;
    const int cBase = blockIdx.y * TN + waveN * 32;

    f32x8 acc[2][2] = {};

    for (int kb = 0; kb < NDIM; kb += KSTEP) {
        bf16x16 ah[2], al[2], bh[2], bl[2];
#pragma unroll
        for (int f = 0; f < 2; ++f) {
            const size_t ra = (size_t)(rBase + f * 16 + ln) * NDIM + kb + half * 8;
            ah[f] = load_frag(Xhi + ra);
            al[f] = load_frag(Xlo + ra);
            const size_t rb = (size_t)(cBase + f * 16 + ln) * NDIM + kb + half * 8;
            bh[f] = load_frag(Xhi + rb);
            bl[f] = load_frag(Xlo + rb);
        }
#pragma unroll
        for (int fm = 0; fm < 2; ++fm) {
#pragma unroll
            for (int fn = 0; fn < 2; ++fn) {
                // split-bf16 fp32 recovery: hi*hi + hi*lo + lo*hi
                acc[fm][fn] = __builtin_amdgcn_wmma_f32_16x16x32_bf16(
                    false, ah[fm], false, bh[fn], (short)0, acc[fm][fn], false, false);
                acc[fm][fn] = __builtin_amdgcn_wmma_f32_16x16x32_bf16(
                    false, ah[fm], false, bl[fn], (short)0, acc[fm][fn], false, false);
                acc[fm][fn] = __builtin_amdgcn_wmma_f32_16x16x32_bf16(
                    false, al[fm], false, bh[fn], (short)0, acc[fm][fn], false, false);
            }
        }
    }

    // L2 = sq_i + sq_j - 2G, clamp >= 0; store + tile partial sum
    float lsum = 0.f;
#pragma unroll
    for (int fm = 0; fm < 2; ++fm) {
#pragma unroll
        for (int fn = 0; fn < 2; ++fn) {
            const int col = cBase + fn * 16 + ln;
            const float sqc = sq[col];
#pragma unroll
            for (int r = 0; r < 8; ++r) {
                const int row = rBase + fm * 16 + half * 8 + r;
                float d = sq[row] + sqc - 2.f * acc[fm][fn][r];
                d = fmaxf(d, 0.f);
                L2[(size_t)row * NROW + col] = d;
                lsum += d;
            }
        }
    }
    // deterministic wave + block reduction of tile sum
    for (int off = 16; off >= 1; off >>= 1)
        lsum += __shfl_xor(lsum, off, 32);
    __shared__ float wred[8];
    if (lane == 0) wred[wave] = lsum;
    __syncthreads();
    if (threadIdx.x == 0) {
        float t = 0.f;
        for (int w = 0; w < 8; ++w) t += wred[w];
        partial[blockIdx.y * gridDim.x + blockIdx.x] = t;
    }
}

// ---------------- K2b: reduce tile partials -> bandwidth ----------------
__global__ void k_bw(const float* __restrict__ partial, int nPart,
                     float* __restrict__ bwOut) {
    __shared__ float red[256];
    const int tid = threadIdx.x;
    float a = 0.f;
    for (int i = tid; i < nPart; i += 256) a += partial[i];
    red[tid] = a;
    __syncthreads();
    for (int s = 128; s > 0; s >>= 1) {
        if (tid < s) red[tid] += red[tid + s];
        __syncthreads();
    }
    if (tid == 0) {
        float b = red[0] / (float)((long long)NROW * NROW - NROW);
        b = b * 0.25f;   // / KERNEL_MUL ** (KERNEL_NUM // 2) = / 4
        bwOut[0] = b;
    }
}

// ---------------- K3: per-row RBF kernels + contrastive stats ----------------
__global__ void k_row(const float* __restrict__ L2, const int* __restrict__ label,
                      const float* __restrict__ bwIn, float* __restrict__ lossArr,
                      float* __restrict__ validArr) {
    const int i = blockIdx.x;
    const int tid = threadIdx.x;
    const float b = bwIn[0];
    float inv[5];
#pragma unroll
    for (int p = 0; p < 5; ++p) inv[p] = 1.0f / (b * (float)(1 << p));
    const float invT = 1.0f / (0.07f * 5.0f);
    const float C = 5.0f * invT;    // rowmax: exact (L2[ii]=0 -> kernels=5)
    const int li = label[i];
    const float* row = L2 + (size_t)i * NROW;

    float s = 0.f, psum = 0.f, pcnt = 0.f;
    for (int j = tid; j < NROW; j += 256) {
        const float d = row[j];
        float k = 0.f;
#pragma unroll
        for (int p = 0; p < 5; ++p) k += __expf(-d * inv[p]);
        const float adc = k * invT;
        if (j != i) {
            s += __expf(adc - C);                       // exp_logits * logits_mask
            if (label[j] == li) { psum += adc; pcnt += 1.f; }
        }
    }
    __shared__ float rs[256], rp[256], rc[256];
    rs[tid] = s; rp[tid] = psum; rc[tid] = pcnt;
    __syncthreads();
    for (int st = 128; st > 0; st >>= 1) {
        if (tid < st) { rs[tid] += rs[tid+st]; rp[tid] += rp[tid+st]; rc[tid] += rc[tid+st]; }
        __syncthreads();
    }
    if (tid == 0) {
        const float S = rs[0], P = rp[0], Cnt = rc[0];
        // sum(mask*log_prob) = (P - Cnt*C) - Cnt*log(S)
        const float mlpp = (P - Cnt * C - Cnt * __logf(S)) / fmaxf(Cnt, 1.f);
        const float loss = -mlpp;                       // T/BASE_T == 1
        const bool valid = Cnt > 0.f;
        lossArr[i]  = valid ? loss : 0.f;
        validArr[i] = valid ? 1.f : 0.f;
    }
}

// ---------------- K4: final scalar ----------------
__global__ void k_final(const float* __restrict__ lossArr,
                        const float* __restrict__ validArr,
                        float* __restrict__ out) {
    __shared__ float rl[256], rv[256];
    const int tid = threadIdx.x;
    float a = 0.f, v = 0.f;
    for (int i = tid; i < NROW; i += 256) { a += lossArr[i]; v += validArr[i]; }
    rl[tid] = a; rv[tid] = v;
    __syncthreads();
    for (int s = 128; s > 0; s >>= 1) {
        if (tid < s) { rl[tid] += rl[tid+s]; rv[tid] += rv[tid+s]; }
        __syncthreads();
    }
    if (tid == 0) out[0] = rl[0] / fmaxf(rv[0], 1.f);
}

extern "C" void kernel_launch(void* const* d_in, const int* in_sizes, int n_in,
                              void* d_out, int out_size, void* d_ws, size_t ws_size,
                              hipStream_t stream) {
    const float* src   = (const float*)d_in[0];
    const int*   label = (const int*)d_in[1];

    char* w = (char*)d_ws;
    // offsets (all 256B aligned)
    __bf16* Xhi     = (__bf16*)(w + 0);                     // 4 MB
    __bf16* Xlo     = (__bf16*)(w + 4194304);               // 4 MB
    float*  sq      = (float*)(w + 8388608);                // 16 KB
    float*  partial = (float*)(w + 8404992);                // 8 KB (2048)
    float*  bw      = (float*)(w + 8413184);                // 256 B
    float*  lossArr = (float*)(w + 8413440);                // 16 KB
    float*  validAr = (float*)(w + 8429824);                // 16 KB
    float*  L2      = (float*)(w + 8446208);                // 64 MB

    k_prep<<<NROW, 256, 0, stream>>>(src, Xhi, Xlo, sq);
    dim3 grid(NROW / TM, NROW / TN);
    k_gemm<<<grid, 256, 0, stream>>>(Xhi, Xlo, sq, L2, partial);
    k_bw<<<1, 256, 0, stream>>>(partial, (NROW / TM) * (NROW / TN), bw);
    k_row<<<NROW, 256, 0, stream>>>(L2, label, bw, lossArr, validAr);
    k_final<<<1, 256, 0, stream>>>(lossArr, validAr, (float*)d_out);
}
